// SharedLSTMDecoder_77996606095469
// MI455X (gfx1250) — matrix operand order repaired and checked by reading, not compiled, not run on hardware
//
#include <hip/hip_runtime.h>
#include <hip/hip_bf16.h>

// CDNA5 / gfx1250 persistent LSTM decoder.
// B=128, H=1024, T=seq_len (400), D = 64 (eeg) or 80 (aud).

typedef __attribute__((ext_vector_type(16))) __bf16 v16bf;
typedef __attribute__((ext_vector_type(8)))  __bf16 v8bf;
typedef __attribute__((ext_vector_type(8)))  float  v8f;

#define NWG   64
#define TPB   256
#define HDIM  1024
#define BDIM  128
#define LDS_ROW  1032                 // halves per LDS row (1024 + 8 pad -> conflict-free)
#define LDS_ROWS 80                   // 16 units x 5 tiles (i,f,g,o,out), interleaved
#define LDS_BYTES (LDS_ROWS * LDS_ROW * 2)

static __device__ __forceinline__ unsigned short f2bf(float f) {
  union { float f; unsigned u; } v; v.f = f;
  unsigned u = v.u;
  u += 0x7fffu + ((u >> 16) & 1u);    // round-to-nearest-even
  return (unsigned short)(u >> 16);
}

// ---- branchless fast activations (avoid ocml's exec-mask branches) ----
static __device__ __forceinline__ float fast_tanh(float x) {
#if __has_builtin(__builtin_amdgcn_tanhf)
  return __builtin_amdgcn_tanhf(x);          // native V_TANH_F32
#elif __has_builtin(__builtin_amdgcn_tanh_f32)
  return __builtin_amdgcn_tanh_f32(x);
#else
  float ax = __builtin_fabsf(x);
  float e  = __expf(-2.f * ax);              // in (0,1], no overflow
  float r  = (1.f - e) * __builtin_amdgcn_rcpf(1.f + e);
  return __builtin_copysignf(r, x);
#endif
}

static __device__ __forceinline__ float fast_sigmoid(float x) {
  return __builtin_amdgcn_rcpf(1.f + __expf(-x));   // exp(-x)->inf => rcp->0, branchless
}

// ---------------- prep kernels ----------------

// x_proj[b][n] = (concat(lat_seg,lat_seq)[b] + mode_emb[mode]) . W_ih[n] + b_ih[n] + b_hh[n]
__global__ void k_xproj(const float* __restrict__ seg, const float* __restrict__ seq,
                        const float* __restrict__ Wih, const float* __restrict__ bih,
                        const float* __restrict__ bhh, const float* __restrict__ memb,
                        const int* __restrict__ modeP, float* __restrict__ X) {
  int idx = blockIdx.x * blockDim.x + threadIdx.x;
  if (idx >= BDIM * 4 * HDIM) return;
  int b = idx >> 12;
  int n = idx & 4095;
  int mode = *modeP;
  const float* w  = Wih + (size_t)n * 64;
  const float* me = memb + mode * 64;
  float acc = bih[n] + bhh[n];
#pragma unroll 8
  for (int i = 0; i < 32; ++i) acc += (seg[b * 32 + i] + me[i]) * w[i];
#pragma unroll 8
  for (int i = 0; i < 32; ++i) acc += (seq[b * 32 + i] + me[32 + i]) * w[32 + i];
  X[idx] = acc;
}

__global__ void k_cvt_whh(const float* __restrict__ Whh, unsigned short* __restrict__ Wg) {
  int idx = blockIdx.x * blockDim.x + threadIdx.x;
  if (idx < 4 * HDIM * HDIM) Wg[idx] = f2bf(Whh[idx]);
}

// Interleave output weights: row 2d = Wmu[d], row 2d+1 = Wlv[d]  (bf16), plus bias vector.
__global__ void k_build_out(const float* __restrict__ WmuE, const float* __restrict__ bmuE,
                            const float* __restrict__ WlvE, const float* __restrict__ blvE,
                            const float* __restrict__ WmuA, const float* __restrict__ bmuA,
                            const float* __restrict__ WlvA, const float* __restrict__ blvA,
                            const int* __restrict__ modeP,
                            unsigned short* __restrict__ Wo, float* __restrict__ bo) {
  int idx = blockIdx.x * blockDim.x + threadIdx.x;
  if (idx >= 160 * HDIM) return;
  int row = idx >> 10, k = idx & (HDIM - 1);
  int mode = *modeP;
  int D = (mode == 0) ? 64 : 80;
  if (row >= 2 * D) return;
  int d = row >> 1, s = row & 1;
  const float* W = (mode == 0) ? (s ? WlvE : WmuE) : (s ? WlvA : WmuA);
  Wo[(size_t)row * HDIM + k] = f2bf(W[(size_t)d * HDIM + k]);
  if (k == 0) {
    const float* bb = (mode == 0) ? (s ? blvE : bmuE) : (s ? blvA : bmuA);
    bo[row] = bb[d];
  }
}

// ---------------- helpers ----------------

static __device__ __forceinline__ v16bf load_afrag(const unsigned short* __restrict__ abase,
                                                   int kk) {
  const unsigned short* ap = abase + kk;   // abase already includes row & half-select offset
  v8bf alo = *(const v8bf*)ap;             // K chunk 0  (contiguous k = 16h + s, matches B)
  v8bf ahi = *(const v8bf*)(ap + 8);       // K chunk 1
  v16bf a;
#pragma unroll
  for (int i = 0; i < 8; ++i) { a[i] = alo[i]; a[8 + i] = ahi[i]; }
  return a;
}

#define WMMA_BF16(A, B, C) \
  __builtin_amdgcn_wmma_f32_16x16x32_bf16(false, (A), false, (B), (short)0, (C), false, false)

// Fully-unrolled k-loop; all B addresses are base + 16-bit immediate thanks to the
// interleaved LDS layout (row = unit*5 + tile).  HASOUT adds the out-proj tile.
template <bool HASOUT>
static __device__ __forceinline__ void gate_gemm(
    const unsigned short* __restrict__ abase, const unsigned short* bbase,
    v8f& acc0, v8f& acc1, v8f& acc2, v8f& acc3, v8f& accO) {
#pragma unroll
  for (int kk = 0; kk < HDIM; kk += 32) {
    v16bf a = load_afrag(abase, kk);
    v16bf b0 = *(const v16bf*)(bbase + 0 * LDS_ROW + kk);
    v16bf b1 = *(const v16bf*)(bbase + 1 * LDS_ROW + kk);
    v16bf b2 = *(const v16bf*)(bbase + 2 * LDS_ROW + kk);
    v16bf b3 = *(const v16bf*)(bbase + 3 * LDS_ROW + kk);
    acc0 = WMMA_BF16(a, b0, acc0);
    acc1 = WMMA_BF16(a, b1, acc1);
    acc2 = WMMA_BF16(a, b2, acc2);
    acc3 = WMMA_BF16(a, b3, acc3);
    if (HASOUT) {
      v16bf b4 = *(const v16bf*)(bbase + 4 * LDS_ROW + kk);
      accO = WMMA_BF16(a, b4, accO);
    }
  }
}

static __device__ __forceinline__ void out_store(v8f accO, const float* __restrict__ bo,
                                                 const float* __restrict__ eps,
                                                 float* __restrict__ out,
                                                 int wg, int lane, int b0, int aoff,
                                                 int t_out, int D, int Tval) {
  int ncol = wg * 16 + (lane & 15);
  float bias = bo[ncol];
  int dd = ncol >> 1;
  bool isMu = (ncol & 1) == 0;
  size_t BTD = (size_t)BDIM * (size_t)Tval * (size_t)D;
#pragma unroll
  for (int r = 0; r < 8; ++r) {
    float own = accO[r] + bias;           // biased mu (even cols) or lv (odd cols)
    float oth = __shfl_xor(own, 1, 32);   // partner column (lv for mu lanes)
    if (isMu) {
      int b = b0 + r + aoff;
      float mu = own, lv = oth;
      float e = eps[(size_t)t_out * BDIM * D + (size_t)b * D + dd];
      float samp = e * __expf(0.5f * lv) + mu;
      size_t o = (size_t)b * Tval * D + (size_t)t_out * D + dd;
      out[o]           = mu;
      out[BTD + o]     = lv;
      out[2 * BTD + o] = samp;
    }
  }
}

// ---------------- main persistent kernel ----------------

__global__ void __launch_bounds__(TPB, 1)
k_lstm(const float* __restrict__ X, const unsigned short* __restrict__ Wg,
       const unsigned short* __restrict__ Wo, const float* __restrict__ bo,
       unsigned short* __restrict__ hbuf, int* __restrict__ cnt,
       const float* __restrict__ eps, const int* __restrict__ modeP,
       const int* __restrict__ seqP, float* __restrict__ out) {
  extern __shared__ unsigned short smem[];

  const int wg   = blockIdx.x;          // owns hidden units [16*wg, 16*wg+16)
  const int tid  = threadIdx.x;
  const int lane = tid & 31;
  const int wave = tid >> 5;            // 8 waves: wave w owns batch tile [16w,16w+16)
  const int b0   = wave * 16;
  const int mode = *modeP;
  const int D    = (mode == 0) ? 64 : 80;
  const int nOut = (2 * D) >> 4;        // 8 (eeg) or 10 (aud) output N-tiles
  const int Tval = *seqP;
  const bool hasOut = (wg < nOut);

  // ---- stage this WG's B-matrix slice into LDS (persists across all timesteps) ----
  // Interleaved layout: LDS row = unit*5 + tile  (tile 0..3 = gates i,f,g,o; 4 = out)
  for (int i = tid; i < LDS_ROWS * (HDIM / 8); i += TPB) {
    int rr = i >> 7;                    // LDS row (0..79)
    int cc = (i & 127) << 3;            // column in halves (step 8)
    int jj = rr / 5;                    // hidden-unit within this WG's slice
    int tile = rr - jj * 5;             // 0..4
    const unsigned short* src;
    if (tile < 4) {
      src = Wg + ((size_t)(tile * HDIM + wg * 16 + jj)) * HDIM + cc;
    } else {
      if (!hasOut) continue;            // out rows only for out-owning WGs
      src = Wo + ((size_t)(wg * 16 + jj)) * HDIM + cc;
    }
    *(uint4*)(&smem[rr * LDS_ROW + cc]) = *(const uint4*)src;
  }
  __syncthreads();

  // ---- preload x_proj tiles (constant across time) into registers ----
  v8f xinit[4];
  {
    int brow = b0 + ((lane < 16) ? 0 : 8);
    int ncl = lane & 15;
#pragma unroll
    for (int g = 0; g < 4; ++g)
#pragma unroll
      for (int r = 0; r < 8; ++r)
        xinit[g][r] = X[(size_t)(brow + r) * (4 * HDIM) + g * HDIM + wg * 16 + ncl];
  }

  float cst[8];                         // cell state, register-resident all 400 steps
#pragma unroll
  for (int r = 0; r < 8; ++r) cst[r] = 0.f;

  const int aoff = (lane < 16) ? 0 : 8;
  const int boff = (lane < 16) ? 0 : 16;
  const int arow = b0 + (lane & 15);
  const unsigned short* bbase = smem + (lane & 15) * (5 * LDS_ROW) + boff;

  for (int t = 0; t < Tval; ++t) {
    const unsigned short* hprev = hbuf + (size_t)(t & 1) * (BDIM * HDIM);
    unsigned short*       hnext = hbuf + (size_t)((t + 1) & 1) * (BDIM * HDIM);
    const unsigned short* abase = hprev + (size_t)arow * HDIM + 2 * aoff;

    v8f acc0 = xinit[0], acc1 = xinit[1], acc2 = xinit[2], acc3 = xinit[3];
    v8f accO = {};

    if (hasOut)
      gate_gemm<true >(abase, bbase, acc0, acc1, acc2, acc3, accO);
    else
      gate_gemm<false>(abase, bbase, acc0, acc1, acc2, acc3, accO);

    // ---- pointwise LSTM cell update for this wave's 16(batch) x 16(unit) tile ----
    unsigned short* hp = hnext + (size_t)(b0 + aoff) * HDIM + wg * 16 + (lane & 15);
#pragma unroll
    for (int r = 0; r < 8; ++r) {
      float ig = fast_sigmoid(acc0[r]);
      float fg = fast_sigmoid(acc1[r]);
      float gg = fast_tanh(acc2[r]);
      float og = fast_sigmoid(acc3[r]);
      float c  = fg * cst[r] + ig * gg;
      cst[r] = c;
      float h = og * fast_tanh(c);
      hp[(size_t)r * HDIM] = f2bf(h);
    }

    // ---- finish pipelined mu/lv/samp for step t-1 ----
    if (hasOut && t > 0)
      out_store(accO, bo, eps, out, wg, lane, b0, aoff, t - 1, D, Tval);

    // ---- one grid-wide barrier per timestep (L2 atomics, monotone counter) ----
    __syncthreads();
    if (tid == 0) {
      __threadfence();
      atomicAdd(cnt, 1);
      int target = (t + 1) * NWG;
      while (atomicAdd(cnt, 0) < target) __builtin_amdgcn_s_sleep(2);
    }
    __syncthreads();
  }

  // ---- epilogue: output projection for the final hidden state h_{T-1} ----
  if (hasOut) {
    const unsigned short* hprev = hbuf + (size_t)(Tval & 1) * (BDIM * HDIM);
    const unsigned short* abase = hprev + (size_t)arow * HDIM + 2 * aoff;
    v8f accO = {};
#pragma unroll
    for (int kk = 0; kk < HDIM; kk += 32) {
      v16bf a = load_afrag(abase, kk);
      v16bf b4 = *(const v16bf*)(bbase + 4 * LDS_ROW + kk);
      accO = WMMA_BF16(a, b4, accO);
    }
    out_store(accO, bo, eps, out, wg, lane, b0, aoff, Tval - 1, D, Tval);
  }
}

// ---------------- launcher ----------------

extern "C" void kernel_launch(void* const* d_in, const int* in_sizes, int n_in,
                              void* d_out, int out_size, void* d_ws, size_t ws_size,
                              hipStream_t stream) {
  (void)in_sizes; (void)n_in; (void)out_size; (void)ws_size;

  const int*   modeP = (const int*)d_in[0];
  const int*   seqP  = (const int*)d_in[1];
  const float* seg   = (const float*)d_in[2];
  const float* seq   = (const float*)d_in[3];
  const float* Wih   = (const float*)d_in[4];
  const float* Whh   = (const float*)d_in[5];
  const float* bih   = (const float*)d_in[6];
  const float* bhh   = (const float*)d_in[7];
  const float* WmuE  = (const float*)d_in[8];
  const float* bmuE  = (const float*)d_in[9];
  const float* WlvE  = (const float*)d_in[10];
  const float* blvE  = (const float*)d_in[11];
  const float* WmuA  = (const float*)d_in[12];
  const float* bmuA  = (const float*)d_in[13];
  const float* WlvA  = (const float*)d_in[14];
  const float* blvA  = (const float*)d_in[15];
  const float* memb  = (const float*)d_in[16];
  const float* eps   = (const float*)d_in[17];
  float* out = (float*)d_out;

  // workspace carve-up
  char* w = (char*)d_ws;
  float*          X    = (float*)(w + 0);                    // 128*4096*4   = 2,097,152
  unsigned short* Wg   = (unsigned short*)(w + 2097152);     // 4096*1024*2  = 8,388,608
  unsigned short* Wo   = (unsigned short*)(w + 10485760);    // 160*1024*2   =   327,680
  float*          bo   = (float*)(w + 10813440);             // 160*4 (pad to 1KB)
  unsigned short* hbuf = (unsigned short*)(w + 10814464);    // 2*128*1024*2 =   524,288
  int*            cnt  = (int*)(w + 11338752);               // barrier counter

  // h0 = 0 and barrier counter = 0 (contiguous region)
  hipMemsetAsync(w + 10814464, 0, 524288 + 64, stream);

  k_xproj<<<(BDIM * 4 * HDIM + 255) / 256, 256, 0, stream>>>(seg, seq, Wih, bih, bhh, memb, modeP, X);
  k_cvt_whh<<<(4 * HDIM * HDIM + 255) / 256, 256, 0, stream>>>(Whh, Wg);
  k_build_out<<<(160 * HDIM + 255) / 256, 256, 0, stream>>>(WmuE, bmuE, WlvE, blvE,
                                                            WmuA, bmuA, WlvA, blvA,
                                                            modeP, Wo, bo);

  k_lstm<<<dim3(NWG), dim3(TPB), LDS_BYTES, stream>>>(X, Wg, Wo, bo, hbuf, cnt,
                                                      eps, modeP, seqP, out);
}
